// TLC_DQN_6914897346938
// MI455X (gfx1250) — compile-verified, matrix operand order checked by reading, not run on hardware
//
#include <hip/hip_runtime.h>

// ---------------------------------------------------------------------------
// TLC-DQN forward for gfx1250 (MI455X). All matmuls via v_wmma_f32_16x16x32_bf16.
// Dominant cost: GRU recurrence (~242 GFLOP) -> persistent per-row-block kernel.
// All LDS staging (GEMM A-tiles, GRU x_t tiles) uses global_load_async_to_lds_b128
// (ASYNCcnt DMA) so fetches overlap WMMA issue; completion via s_wait_asynccnt.
// ---------------------------------------------------------------------------

typedef __bf16 bf16;
typedef bf16  v16bf __attribute__((ext_vector_type(16)));
typedef bf16  v8bf  __attribute__((ext_vector_type(8)));
typedef float v8f   __attribute__((ext_vector_type(8)));

__device__ __forceinline__ bf16 f2bf(float f) {
  unsigned u = __builtin_bit_cast(unsigned, f);
  unsigned r = u + 0x7FFFu + ((u >> 16) & 1u);          // round-to-nearest-even
  unsigned short h = (unsigned short)(r >> 16);
  return __builtin_bit_cast(bf16, h);
}

__device__ __forceinline__ v8f vzero8() {
  v8f z;
#pragma unroll
  for (int i = 0; i < 8; ++i) z[i] = 0.f;
  return z;
}

// 16B async DMA: global -> LDS, tracked by ASYNCcnt.
__device__ __forceinline__ void async_b128(const void* lds_dst, const void* gsrc) {
  asm volatile("global_load_async_to_lds_b128 %0, %1, off"
               :: "v"((unsigned)(size_t)lds_dst),
                  "v"((unsigned long long)(size_t)gsrc)
               : "memory");
}
__device__ __forceinline__ void wait_async0() {
  asm volatile("s_wait_asynccnt 0" ::: "memory");
}

// A-fragment (16x32 bf16, row-major in LDS, leading dim = ld).
// Per ISA: lanes 0-15 -> K 0..7 (V0..3) & 16..23 (V4..7); lanes 16-31 -> +8.
__device__ __forceinline__ v16bf load_a_frag(const bf16* base, int ld) {
  const int lane = threadIdx.x & 31;
  const bf16* p = base + (lane & 15) * ld + ((lane >> 4) << 3);
  v8bf lo = *(const v8bf*)(p);
  v8bf hi = *(const v8bf*)(p + 16);
  v16bf r;
#pragma unroll
  for (int i = 0; i < 8; ++i) { r[i] = lo[i]; r[8 + i] = hi[i]; }
  return r;
}

// B-fragment: pre-packed at weight-prep time into 32B/lane contiguous chunks.
__device__ __forceinline__ v16bf load_b_frag(const bf16* frag) {
  const int lane = threadIdx.x & 31;
  return *(const v16bf*)(frag + lane * 16);
}

__device__ __forceinline__ v8f wmma_bf16(v16bf a, v16bf b, v8f c) {
  return __builtin_amdgcn_wmma_f32_16x16x32_bf16(false, a, false, b, (short)0, c,
                                                 false, false);
}

__device__ __forceinline__ float sigmoidf_(float x) { return 1.f / (1.f + expf(-x)); }

// ---------------------------------------------------------------------------
// Pack a [K,N] (or transposed [N,K]) fp32 weight into WMMA B-fragment order.
// frag id = nt*ktiles_total + (kt_off+ktl); each frag = 32 lanes * 16 bf16.
// ---------------------------------------------------------------------------
__global__ void pack_b_kernel(const float* __restrict__ src, bf16* __restrict__ dst,
                              int Ksrc, int N, int ld, int trans, int kt_off,
                              int ktiles_total) {
  const int ktiles_src = (Ksrc + 31) >> 5;
  const int ntiles = N >> 4;
  const long long total = (long long)ktiles_src * ntiles * 32;
  for (long long i = (long long)blockIdx.x * 256 + threadIdx.x; i < total;
       i += (long long)gridDim.x * 256) {
    const int lane = (int)(i & 31);
    const int fr = (int)(i >> 5);
    const int nt = fr % ntiles;
    const int ktl = fr / ntiles;
    bf16* d = dst + ((size_t)nt * ktiles_total + kt_off + ktl) * 512 + lane * 16;
    const int n = nt * 16 + (lane & 15);
    const int grp = lane >> 4;
#pragma unroll
    for (int e = 0; e < 16; ++e) {
      const int j = e >> 1, half = e & 1;
      int k;
      if (j < 4) k = ktl * 32 + grp * 8 + 2 * j + half;
      else       k = ktl * 32 + grp * 8 + 16 + 2 * (j - 4) + half;
      float v = 0.f;
      if (k < Ksrc) v = trans ? src[(size_t)n * ld + k] : src[(size_t)k * ld + n];
      d[e] = f2bf(v);
    }
  }
}

// ---------------------------------------------------------------------------
// Generic WMMA GEMM: C = act(A @ B + bias). A bf16 [M,K] row-major, K%32==0,
// M%64==0, N%32==0 (all call sites satisfy this). B packed fragments. C fp32 or
// bf16 with leading dim ldc. Block = 256 thr = 8 waves covering BM=64 x BN=32
// (one 16x16 tile per wave). LDS double-buffered; next A chunk is DMA'd with
// global_load_async_to_lds_b128 while the current buffer feeds the WMMA.
// ---------------------------------------------------------------------------
__global__ __launch_bounds__(256) void wmma_gemm_kernel(
    const bf16* __restrict__ A, const bf16* __restrict__ Bp,
    const float* __restrict__ bias, void* __restrict__ Cout, int M, int N, int K,
    int ldc, int relu, int out_bf16) {
  const int ktiles = K >> 5;
  __shared__ __align__(16) bf16 As[2][64 * 32];
  const int wave = threadIdx.x >> 5;
  const int lane = threadIdx.x & 31;
  const int mt = wave & 3;
  const int nt = blockIdx.y * 2 + (wave >> 2);
  const int m0 = blockIdx.x * 64;

  // staging: each thread owns one 16B chunk of the 64x32 tile
  const int srow = threadIdx.x >> 2;         // 0..63
  const int scol = (threadIdx.x & 3) << 3;   // 0,8,16,24 (bf16 elements)
  const bf16* gsrc = A + (size_t)(m0 + srow) * K + scol;
  bf16* sd0 = As[0] + srow * 32 + scol;
  bf16* sd1 = As[1] + srow * 32 + scol;
  async_b128(sd0, gsrc);
  wait_async0();
  __syncthreads();

  v8f acc = vzero8();
  for (int kt = 0; kt < ktiles; ++kt) {
    if (kt + 1 < ktiles)                      // DMA next K-chunk into alt buffer
      async_b128((kt & 1) ? sd0 : sd1, gsrc + (kt + 1) * 32);
    v16bf a = load_a_frag(As[kt & 1] + mt * 16 * 32, 32);
    v16bf b = load_b_frag(Bp + ((size_t)nt * ktiles + kt) * 512);
    acc = wmma_bf16(a, b, acc);
    wait_async0();
    __syncthreads();
  }
  {
    const int n = nt * 16 + (lane & 15);
    const int grp = lane >> 4;
    const float bi = bias ? bias[n] : 0.f;
#pragma unroll
    for (int j = 0; j < 8; ++j) {
      const int gm = m0 + mt * 16 + j + 8 * grp;
      float v = acc[j] + bi;
      if (relu) v = fmaxf(v, 0.f);
      if (out_bf16) ((bf16*)Cout)[(size_t)gm * ldc + n] = f2bf(v);
      else          ((float*)Cout)[(size_t)gm * ldc + n] = v;
    }
  }
}

// ---------------------------------------------------------------------------
// Gathers / im2col (elementwise, grid-stride)
// ---------------------------------------------------------------------------
__global__ void gather_mlp_kernel(const float* __restrict__ obs, bf16* __restrict__ A) {
  const long long total = 307200LL * 32;  // rows = (b*24+j)*50+t, K padded 16->32
  for (long long i = (long long)blockIdx.x * 256 + threadIdx.x; i < total;
       i += (long long)gridDim.x * 256) {
    const int k = (int)(i & 31);
    const long long row = i >> 5;
    const int t = (int)(row % 50);
    const long long seq = row / 50;
    const int b = (int)(seq / 24), j = (int)(seq % 24);
    float v = 0.f;
    if (k < 16) v = obs[(((size_t)b * 25 + 1 + j) * 50 + t) * 16 + k];
    A[i] = f2bf(v);
  }
}

__global__ void im2col_c1_kernel(const float* __restrict__ obs_map, bf16* __restrict__ A) {
  const long long total = 102400LL * 192;  // rows = b*400 + oy*20+ox
  for (long long i = (long long)blockIdx.x * 256 + threadIdx.x; i < total;
       i += (long long)gridDim.x * 256) {
    const int k = (int)(i % 192);
    const long long row = i / 192;
    const int pos = (int)(row % 400), b = (int)(row / 400);
    const int oy = pos / 20, ox = pos % 20;
    const int c = k / 64, rem = k % 64, kh = rem / 8, kw = rem % 8;
    A[i] = f2bf(obs_map[(((size_t)b * 3 + c) * 84 + oy * 4 + kh) * 84 + ox * 4 + kw]);
  }
}

__global__ void im2col_c2_kernel(const bf16* __restrict__ src, bf16* __restrict__ A) {
  const long long total = 20736LL * 512;  // src NHWC [b*400+y*20+x, 32]
  for (long long i = (long long)blockIdx.x * 256 + threadIdx.x; i < total;
       i += (long long)gridDim.x * 256) {
    const int k = (int)(i % 512);
    const long long row = i / 512;
    const int pos = (int)(row % 81), b = (int)(row / 81);
    const int oy = pos / 9, ox = pos % 9;
    const int c = k >> 4, rem = k & 15, kh = rem >> 2, kw = rem & 3;
    A[i] = src[((size_t)b * 400 + (oy * 2 + kh) * 20 + (ox * 2 + kw)) * 32 + c];
  }
}

__global__ void im2col_c3_kernel(const bf16* __restrict__ src, bf16* __restrict__ A) {
  const long long total = 12544LL * 576;  // src NHWC [b*81+y*9+x, 64]
  for (long long i = (long long)blockIdx.x * 256 + threadIdx.x; i < total;
       i += (long long)gridDim.x * 256) {
    const int k = (int)(i % 576);
    const long long row = i / 576;
    const int pos = (int)(row % 49), b = (int)(row / 49);
    const int oy = pos / 7, ox = pos % 7;
    const int c = k / 9, rem = k % 9, kh = rem / 3, kw = rem % 3;
    A[i] = src[((size_t)b * 81 + (oy + kh) * 9 + (ox + kw)) * 64 + c];
  }
}

__global__ void gather_fc_kernel(const bf16* __restrict__ src, bf16* __restrict__ A) {
  const long long total = 256LL * 3136;  // NCHW flatten from NHWC conv3 out
  for (long long i = (long long)blockIdx.x * 256 + threadIdx.x; i < total;
       i += (long long)gridDim.x * 256) {
    const int k = (int)(i % 3136);
    const int b = (int)(i / 3136);
    const int c = k / 49, rem = k % 49, y = rem / 7, x = rem % 7;
    A[i] = src[((size_t)b * 49 + y * 7 + x) * 64 + c];
  }
}

// ---------------------------------------------------------------------------
// Persistent GRU. Block owns 32 sequences (2 M-tiles); 8 waves each own a
// 32-column slice of the hidden dim (2 col-tiles per gate). Gate GEMM per step:
// [32,512] x [512,768] with K split x-part (nbrs_emb via async-to-LDS) /
// h-part (bf16 LDS mirror). r,z accumulate x+h jointly; n-gate keeps x/h
// separate (n = tanh(xn + r*hn)). h kept fp32 in registers. x_{t+1} is DMA'd
// with global_load_async_to_lds_b128 (ASYNCcnt) while the gate math runs.
// ---------------------------------------------------------------------------
__global__ __launch_bounds__(256) void gru_kernel(
    const bf16* __restrict__ xemb,   // [6144*50, 256] relu'd MLP output
    const bf16* __restrict__ Wp,     // packed [K=512, N=768], frag = nt*16+kt
    const float* __restrict__ bih, const float* __restrict__ bhh,
    bf16* __restrict__ node_emb) {   // [256*25, 256]
  __shared__ __align__(16) bf16 Ax[2][32 * 256];
  __shared__ __align__(16) bf16 Ah[32 * 256];
  const int tid = threadIdx.x;
  const int wave = tid >> 5, lane = tid & 31;
  const int laneN = lane & 15, laneG = lane >> 4;
  const int s0 = blockIdx.x * 32;
  const int c0 = wave * 32;

  // async DMA of one timestep's x block (32 rows x 256 bf16 = 1024 x 16B)
  auto stage_async = [&](int buf, int t) {
#pragma unroll
    for (int i = tid; i < 1024; i += 256) {
      const int r = i >> 5, c = i & 31;
      async_b128(&Ax[buf][r * 256 + (c << 3)],
                 xemb + (size_t)((s0 + r) * 50 + t) * 256 + (c << 3));
    }
  };

  float br[2], bz[2], bnx[2], bnh[2];
#pragma unroll
  for (int ct = 0; ct < 2; ++ct) {
    const int n = c0 + ct * 16 + laneN;
    br[ct]  = bih[n] + bhh[n];
    bz[ct]  = bih[256 + n] + bhh[256 + n];
    bnx[ct] = bih[512 + n];
    bnh[ct] = bhh[512 + n];
  }

  float hreg[2][2][8];
#pragma unroll
  for (int mt = 0; mt < 2; ++mt)
#pragma unroll
    for (int ct = 0; ct < 2; ++ct)
#pragma unroll
      for (int j = 0; j < 8; ++j) hreg[mt][ct][j] = 0.f;

  const bf16 zb = f2bf(0.f);
  for (int i = tid; i < 32 * 256; i += 256) Ah[i] = zb;
  stage_async(0, 0);
  wait_async0();
  __syncthreads();

  for (int t = 0; t < 50; ++t) {
    const bf16* Axc = Ax[t & 1];

    v8f Cr[2][2], Cz[2][2], Cnx[2][2], Cnh[2][2];
#pragma unroll
    for (int mt = 0; mt < 2; ++mt)
#pragma unroll
      for (int ct = 0; ct < 2; ++ct) {
        Cr[mt][ct] = vzero8(); Cz[mt][ct] = vzero8();
        Cnx[mt][ct] = vzero8(); Cnh[mt][ct] = vzero8();
      }

    const int nr = 2 * wave, nz = 16 + 2 * wave, nn = 32 + 2 * wave;
    for (int kt = 0; kt < 16; ++kt) {
      const bf16* Asrc = (kt < 8) ? Axc : Ah;
      const int ktl = kt & 7;
      v16bf Br0 = load_b_frag(Wp + ((size_t)(nr + 0) * 16 + kt) * 512);
      v16bf Br1 = load_b_frag(Wp + ((size_t)(nr + 1) * 16 + kt) * 512);
      v16bf Bz0 = load_b_frag(Wp + ((size_t)(nz + 0) * 16 + kt) * 512);
      v16bf Bz1 = load_b_frag(Wp + ((size_t)(nz + 1) * 16 + kt) * 512);
      v16bf Bn0 = load_b_frag(Wp + ((size_t)(nn + 0) * 16 + kt) * 512);
      v16bf Bn1 = load_b_frag(Wp + ((size_t)(nn + 1) * 16 + kt) * 512);
#pragma unroll
      for (int mt = 0; mt < 2; ++mt) {
        v16bf a = load_a_frag(Asrc + mt * 16 * 256 + ktl * 32, 256);
        Cr[mt][0] = wmma_bf16(a, Br0, Cr[mt][0]);
        Cr[mt][1] = wmma_bf16(a, Br1, Cr[mt][1]);
        Cz[mt][0] = wmma_bf16(a, Bz0, Cz[mt][0]);
        Cz[mt][1] = wmma_bf16(a, Bz1, Cz[mt][1]);
        if (kt < 8) {
          Cnx[mt][0] = wmma_bf16(a, Bn0, Cnx[mt][0]);
          Cnx[mt][1] = wmma_bf16(a, Bn1, Cnx[mt][1]);
        } else {
          Cnh[mt][0] = wmma_bf16(a, Bn0, Cnh[mt][0]);
          Cnh[mt][1] = wmma_bf16(a, Bn1, Cnh[mt][1]);
        }
      }
    }
    __syncthreads();  // everyone done reading Ah / Ax[t&1]

    if (t + 1 < 50) stage_async((t + 1) & 1, t + 1);  // DMA overlaps gate math

#pragma unroll
    for (int mt = 0; mt < 2; ++mt)
#pragma unroll
      for (int ct = 0; ct < 2; ++ct) {
        const int n = c0 + ct * 16 + laneN;
#pragma unroll
        for (int j = 0; j < 8; ++j) {
          const float r = sigmoidf_(Cr[mt][ct][j] + br[ct]);
          const float z = sigmoidf_(Cz[mt][ct][j] + bz[ct]);
          const float ng =
              tanhf(Cnx[mt][ct][j] + bnx[ct] + r * (Cnh[mt][ct][j] + bnh[ct]));
          const float h = (1.f - z) * ng + z * hreg[mt][ct][j];
          hreg[mt][ct][j] = h;
          const int m = mt * 16 + j + 8 * laneG;
          Ah[m * 256 + n] = f2bf(h);
        }
      }
    wait_async0();
    __syncthreads();  // Ah + Ax[(t+1)&1] ready for next step
  }

  // final hidden -> node_emb rows b*25 + 1 + j
#pragma unroll
  for (int mt = 0; mt < 2; ++mt)
#pragma unroll
    for (int ct = 0; ct < 2; ++ct) {
      const int n = c0 + ct * 16 + laneN;
#pragma unroll
      for (int j = 0; j < 8; ++j) {
        const int seq = s0 + mt * 16 + j + 8 * laneG;
        const int b = seq / 24, jn = seq - b * 24;
        node_emb[((size_t)b * 25 + 1 + jn) * 256 + n] = f2bf(hreg[mt][ct][j]);
      }
    }
}

// ---------------------------------------------------------------------------
// GATv2 edge scores + per-dst softmax + node-0 aggregation (one block / graph).
// Only gat_out[:,0] is consumed downstream; alpha[:, :101].mean(heads) -> att_w.
// ---------------------------------------------------------------------------
__global__ __launch_bounds__(256) void gat_edge_kernel(
    const int* __restrict__ edges, const float* __restrict__ Gl,
    const float* __restrict__ Gr, const float* __restrict__ att,
    const float* __restrict__ gbias, float* __restrict__ att_out,
    bf16* __restrict__ out0) {
  __shared__ float sc[625], al[625];
  __shared__ float mx[125], den[125];  // indexed node*5+head
  __shared__ int ssrc[125], sdst[125];
  const int b = blockIdx.x, tid = threadIdx.x;
  const int* eb = edges + (size_t)b * 200;
  for (int e = tid; e < 125; e += 256) {
    if (e < 100) { ssrc[e] = eb[e]; sdst[e] = eb[100 + e]; }
    else         { ssrc[e] = e - 100; sdst[e] = e - 100; }
  }
  __syncthreads();
  const float* glb = Gl + (size_t)b * 25 * 1280;
  const float* grb = Gr + (size_t)b * 25 * 1280;
  for (int task = tid; task < 625; task += 256) {
    const int e = task / 5, hd = task % 5;
    const float* gl = glb + (size_t)ssrc[e] * 1280 + hd * 256;
    const float* gr = grb + (size_t)sdst[e] * 1280 + hd * 256;
    const float* at = att + hd * 256;
    float acc = 0.f;
    for (int c = 0; c < 256; ++c) {
      float v = gl[c] + gr[c];
      v = (v > 0.f) ? v : 0.2f * v;  // leaky_relu(., 0.2)
      acc += v * at[c];
    }
    sc[task] = acc;
  }
  __syncthreads();
  for (int task = tid; task < 125; task += 256) {
    const int node = task / 5, hd = task % 5;
    float m = -1e30f;
    for (int e = 0; e < 125; ++e)
      if (sdst[e] == node) m = fmaxf(m, sc[e * 5 + hd]);
    float d = 0.f;
    for (int e = 0; e < 125; ++e)
      if (sdst[e] == node) d += expf(sc[e * 5 + hd] - m);
    mx[task] = m; den[task] = d;
  }
  __syncthreads();
  for (int task = tid; task < 625; task += 256) {
    const int e = task / 5, hd = task % 5;
    const int idx = sdst[e] * 5 + hd;
    al[task] = expf(sc[task] - mx[idx]) / den[idx];
  }
  __syncthreads();
  for (int e = tid; e < 101; e += 256) {
    float s = 0.f;
    for (int hd = 0; hd < 5; ++hd) s += al[e * 5 + hd];
    att_out[(size_t)b * 101 + e] = s * 0.2f;
  }
  {  // node-0 aggregation: mean over heads of sum_{dst==0} alpha*gl[src]
    const int c = tid;
    float acc = 0.f;
    for (int e = 0; e < 125; ++e) {
      if (sdst[e] == 0) {
        const float* gl = glb + (size_t)ssrc[e] * 1280;
        for (int hd = 0; hd < 5; ++hd) acc += al[e * 5 + hd] * gl[hd * 256 + c];
      }
    }
    out0[(size_t)b * 256 + c] = f2bf(acc * 0.2f + gbias[c]);
  }
}

// ---------------------------------------------------------------------------
// Dueling head: v/adv dots + q = v + adv - mean(adv). One block per batch row.
// ---------------------------------------------------------------------------
__global__ __launch_bounds__(256) void head_kernel(
    const float* __restrict__ hvec, const float* __restrict__ out_w,
    const float* __restrict__ out_b, const float* __restrict__ adv_w,
    const float* __restrict__ adv_b, float* __restrict__ q) {
  __shared__ float red[256];
  __shared__ float s[9];
  const int b = blockIdx.x, t = threadIdx.x;
  const float hv = hvec[(size_t)b * 256 + t];
  for (int o = 0; o < 9; ++o) {
    const float w = (o == 0) ? out_w[t] : adv_w[(size_t)t * 8 + (o - 1)];
    red[t] = hv * w;
    __syncthreads();
    for (int sft = 128; sft > 0; sft >>= 1) {
      if (t < sft) red[t] += red[t + sft];
      __syncthreads();
    }
    if (t == 0) s[o] = red[0];
    __syncthreads();
  }
  if (t == 0) {
    const float v = s[0] + out_b[0];
    float mean = 0.f;
    for (int o = 0; o < 8; ++o) mean += s[1 + o] + adv_b[o];
    mean *= 0.125f;
    for (int o = 0; o < 8; ++o)
      q[(size_t)b * 8 + o] = v + (s[1 + o] + adv_b[o]) - mean;
  }
}

// ---------------------------------------------------------------------------
extern "C" void kernel_launch(void* const* d_in, const int* in_sizes, int n_in,
                              void* d_out, int out_size, void* d_ws, size_t ws_size,
                              hipStream_t stream) {
  (void)in_sizes; (void)n_in; (void)out_size; (void)ws_size;
  const float* obs      = (const float*)d_in[0];
  const float* obs_map  = (const float*)d_in[1];
  const int*   edges    = (const int*)d_in[2];
  // d_in[3] edges_feature: unused by the reference
  const float* mlp_w = (const float*)d_in[4];
  const float* mlp_b = (const float*)d_in[5];
  const float* c1_w = (const float*)d_in[6];  const float* c1_b = (const float*)d_in[7];
  const float* c2_w = (const float*)d_in[8];  const float* c2_b = (const float*)d_in[9];
  const float* c3_w = (const float*)d_in[10]; const float* c3_b = (const float*)d_in[11];
  const float* fc_w = (const float*)d_in[12]; const float* fc_b = (const float*)d_in[13];
  const float* gru_wih = (const float*)d_in[14];
  const float* gru_whh = (const float*)d_in[15];
  const float* gru_bih = (const float*)d_in[16];
  const float* gru_bhh = (const float*)d_in[17];
  const float* gat_wl = (const float*)d_in[18]; const float* gat_bl = (const float*)d_in[19];
  const float* gat_wr = (const float*)d_in[20]; const float* gat_br = (const float*)d_in[21];
  const float* gat_att = (const float*)d_in[22]; const float* gat_bias = (const float*)d_in[23];
  const float* hid_w = (const float*)d_in[24]; const float* hid_b = (const float*)d_in[25];
  const float* out_w = (const float*)d_in[26]; const float* out_b = (const float*)d_in[27];
  const float* adv_w = (const float*)d_in[28]; const float* adv_b = (const float*)d_in[29];

  char* base = (char*)d_ws;
  size_t off = 0;
  auto carve = [&](size_t bytes) -> char* {
    char* p = base + off;
    off += (bytes + 255) & ~(size_t)255;
    return p;
  };
  // packed weights (frag = 1KB)
  bf16* Wmlp = (bf16*)carve((size_t)16 * 1 * 1024);
  bf16* Wgru = (bf16*)carve((size_t)48 * 16 * 1024);
  bf16* Wgl  = (bf16*)carve((size_t)80 * 8 * 1024);
  bf16* Wgr  = (bf16*)carve((size_t)80 * 8 * 1024);
  bf16* Whid = (bf16*)carve((size_t)16 * 8 * 1024);
  bf16* Wc1  = (bf16*)carve((size_t)2 * 6 * 1024);
  bf16* Wc2  = (bf16*)carve((size_t)4 * 16 * 1024);
  bf16* Wc3  = (bf16*)carve((size_t)4 * 18 * 1024);
  bf16* Wfc  = (bf16*)carve((size_t)16 * 98 * 1024);
  // activations
  bf16* A_mlp    = (bf16*)carve(307200ULL * 32 * 2);
  bf16* nbrs_emb = (bf16*)carve(307200ULL * 256 * 2);
  bf16* A_c1     = (bf16*)carve(102400ULL * 192 * 2);
  bf16* conv1out = (bf16*)carve(102400ULL * 32 * 2);
  bf16* A_c2     = (bf16*)carve(20736ULL * 512 * 2);
  bf16* conv2out = (bf16*)carve(20736ULL * 64 * 2);
  bf16* A_c3     = (bf16*)carve(12544ULL * 576 * 2);
  bf16* conv3out = (bf16*)carve(12544ULL * 64 * 2);
  bf16* A_fc     = (bf16*)carve(256ULL * 3136 * 2);
  bf16* node_emb = (bf16*)carve(6400ULL * 256 * 2);
  float* Gl      = (float*)carve(6400ULL * 1280 * 4);
  float* Gr      = (float*)carve(6400ULL * 1280 * 4);
  bf16* out0bf   = (bf16*)carve(256ULL * 256 * 2);
  float* hvec    = (float*)carve(256ULL * 256 * 4);

  float* q_out   = (float*)d_out;        // [256, 8]
  float* att_out = q_out + 256 * 8;      // [256, 101]

  auto pk = [&](const float* src, bf16* dst, int Ksrc, int N, int ld, int trans,
                int kt_off, int kt_total) {
    const int total = ((Ksrc + 31) >> 5) * (N >> 4) * 32;
    pack_b_kernel<<<(total + 255) / 256, 256, 0, stream>>>(src, dst, Ksrc, N, ld,
                                                           trans, kt_off, kt_total);
  };
  pk(mlp_w,   Wmlp, 16,   256,  256,  0, 0, 1);
  pk(gru_wih, Wgru, 256,  768,  256,  1, 0, 16);   // B[k][n]=wih[n][k]
  pk(gru_whh, Wgru, 256,  768,  256,  1, 8, 16);   // k-tiles 8..15
  pk(gat_wl,  Wgl,  256,  1280, 1280, 0, 0, 8);
  pk(gat_wr,  Wgr,  256,  1280, 1280, 0, 0, 8);
  pk(hid_w,   Whid, 256,  256,  256,  0, 0, 8);
  pk(c1_w,    Wc1,  192,  32,   192,  1, 0, 6);    // OIHW -> B[k][n]=w[n*K+k]
  pk(c2_w,    Wc2,  512,  64,   512,  1, 0, 16);
  pk(c3_w,    Wc3,  576,  64,   576,  1, 0, 18);
  pk(fc_w,    Wfc,  3136, 256,  256,  0, 0, 98);

  auto gemm = [&](const bf16* A, const bf16* Bp, const float* bias, void* C,
                  int M, int N, int K, int ldc, int relu, int obf16) {
    dim3 grid((M + 63) / 64, N / 32);
    wmma_gemm_kernel<<<grid, 256, 0, stream>>>(A, Bp, bias, C, M, N, K, ldc, relu,
                                               obf16);
  };

  // neighbor MLP: [307200,32] x [32,256] -> relu -> bf16
  gather_mlp_kernel<<<2048, 256, 0, stream>>>(obs, A_mlp);
  gemm(A_mlp, Wmlp, mlp_b, nbrs_emb, 307200, 256, 32, 256, 1, 1);

  // Nature-CNN via im2col + GEMM
  im2col_c1_kernel<<<2048, 256, 0, stream>>>(obs_map, A_c1);
  gemm(A_c1, Wc1, c1_b, conv1out, 102400, 32, 192, 32, 1, 1);
  im2col_c2_kernel<<<2048, 256, 0, stream>>>(conv1out, A_c2);
  gemm(A_c2, Wc2, c2_b, conv2out, 20736, 64, 512, 64, 1, 1);
  im2col_c3_kernel<<<2048, 256, 0, stream>>>(conv2out, A_c3);
  gemm(A_c3, Wc3, c3_b, conv3out, 12544, 64, 576, 64, 1, 1);
  gather_fc_kernel<<<1024, 256, 0, stream>>>(conv3out, A_fc);
  // fc writes target_emb straight into node_emb rows b*25 (ldc = 25*256)
  gemm(A_fc, Wfc, fc_b, node_emb, 256, 256, 3136, 6400, 1, 1);

  // persistent GRU (fills node_emb rows b*25 + 1..24)
  gru_kernel<<<192, 256, 0, stream>>>(nbrs_emb, Wgru, gru_bih, gru_bhh, node_emb);

  // GATv2 projections: [6400,256] x [256,1280] (+bias) -> f32
  gemm(node_emb, Wgl, gat_bl, Gl, 6400, 1280, 256, 1280, 0, 0);
  gemm(node_emb, Wgr, gat_br, Gr, 6400, 1280, 256, 1280, 0, 0);

  // edge scores / softmax / node-0 aggregation / att_w
  gat_edge_kernel<<<256, 256, 0, stream>>>(edges, Gl, Gr, gat_att, gat_bias,
                                           att_out, out0bf);

  // hidden head: relu(out0 @ hid_w + hid_b)
  gemm(out0bf, Whid, hid_b, hvec, 256, 256, 256, 256, 1, 0);

  // dueling Q head
  head_kernel<<<256, 256, 0, stream>>>(hvec, out_w, out_b, adv_w, adv_b, q_out);
}
